// Net_19378892440061
// MI455X (gfx1250) — compile-verified, matrix operand order checked by reading, not run on hardware
//
#include <hip/hip_runtime.h>
#include <cstddef>
#include <cstdint>

typedef __attribute__((ext_vector_type(2)))  _Float16 v2h;
typedef __attribute__((ext_vector_type(4)))  _Float16 v4h;
typedef __attribute__((ext_vector_type(8)))  _Float16 v8h;
typedef __attribute__((ext_vector_type(16))) _Float16 v16h;
typedef __attribute__((ext_vector_type(8)))  float    v8f;

template<int N> struct HVec;
template<> struct HVec<4> { typedef v4h T; };
template<> struct HVec<8> { typedef v8h T; };

#define N_FLUID 16000
#define N_WALL  8000
#define KNBR    64
#define NPAIRS  (N_FLUID * KNBR)
#define RADIUS_F 0.1125f   /* 0.5 * 6.0 * 1.5 * 0.025 */

// KPAD: 64*CINP conv block + CINP identity block + 1 bias slot, rounded up to
// a multiple of 256 halves so KPAD/32 is divisible by any split-K factor (<=8).
__host__ __device__ constexpr int kpad_of(int cinp) {
  return ((64 * cinp + cinp + 1 + 255) / 256) * 256;
}

// ---------------------------------------------------------------------------
// Kernel 1: per-(particle,neighbor) geometry -> 8 f16 trilinear*window weights
// plus packed base cell (x*16+y*4+z), 0xFFFFFFFF when pair contributes nothing.
// ---------------------------------------------------------------------------
__global__ __launch_bounds__(256) void geom_kernel(
    const float* __restrict__ pin, const float* __restrict__ pout,
    const int* __restrict__ nbr, const unsigned char* __restrict__ mask,
    int npairs, v8h* __restrict__ pairW, unsigned* __restrict__ pairCell)
{
  int i = blockIdx.x * 256 + threadIdx.x;
  if (i >= npairs) return;
  int n  = i >> 6;
  int nb = nbr[i];
  bool mk = mask[i] != 0;

  const float invR = 1.0f / RADIUS_F;
  float x = (pin[nb * 3 + 0] - pout[n * 3 + 0]) * invR;
  float y = (pin[nb * 3 + 1] - pout[n * 3 + 1]) * invR;
  float z = (pin[nb * 3 + 2] - pout[n * 3 + 2]) * invR;
  if (!mk) { x = 0.f; y = 0.f; z = 0.f; }

  float r2  = x * x + y * y + z * z;
  float t   = 1.f - r2;
  float win = fminf(fmaxf(t * t * t, 0.f), 1.f);
  if (!mk) win = 0.f;

  // ball -> cube (volume preserving), faithful to reference math
  const float eps = 1e-12f;
  float sq   = r2;
  float nrm  = sqrtf(fmaxf(sq, eps));
  float rho  = sqrtf(fmaxf(x * x + y * y, eps));
  bool  top  = (1.25f * z * z) > (x * x + y * y);
  float stp  = sqrtf(3.f * nrm / (nrm + fabsf(z)));
  float xs   = top ? x * stp : x * nrm / rho;
  float ys   = top ? y * stp : y * nrm / rho;
  float sgz  = (z > 0.f) ? 1.f : ((z < 0.f) ? -1.f : 0.f);
  float zs   = top ? sgz * nrm : 1.5f * z;
  if (sq < eps) { xs = 0.f; ys = 0.f; zs = 0.f; }

  float rxy  = sqrtf(fmaxf(xs * xs + ys * ys, eps));
  bool  usex = fabsf(ys) <= fabsf(xs);
  float dx   = (usex && fabsf(xs) > eps) ? xs : 1.f;
  float dy   = (!usex && fabsf(ys) > eps) ? ys : 1.f;
  const float c4pi = 4.f / 3.14159265358979323846f;
  float sxs = (xs > 0.f) ? 1.f : ((xs < 0.f) ? -1.f : 0.f);
  float sys = (ys > 0.f) ? 1.f : ((ys < 0.f) ? -1.f : 0.f);
  float xc = usex ? sxs * rxy : sys * rxy * c4pi * atanf(xs / dy);
  float yc = usex ? sxs * rxy * c4pi * atanf(ys / dx) : sys * rxy;
  if (xs * xs + ys * ys < eps) { xc = 0.f; yc = 0.f; }

  // trilinear coords on 4x4x4 grid
  float cx = fminf(fmaxf((xc * 0.5f + 0.5f) * 3.f, 0.f), 3.f);
  float cy = fminf(fmaxf((yc * 0.5f + 0.5f) * 3.f, 0.f), 3.f);
  float cz = fminf(fmaxf((zs * 0.5f + 0.5f) * 3.f, 0.f), 3.f);
  float ix = fminf(floorf(cx), 2.f);
  float iy = fminf(floorf(cy), 2.f);
  float iz = fminf(floorf(cz), 2.f);
  float fx = cx - ix, fy = cy - iy, fz = cz - iz;
  unsigned base = (unsigned)((int)ix * 16 + (int)iy * 4 + (int)iz);
  float wx0 = 1.f - fx, wy0 = 1.f - fy, wz0 = 1.f - fz;

  v8h wv;
  wv[0] = (_Float16)(win * wx0 * wy0 * wz0);
  wv[1] = (_Float16)(win * wx0 * wy0 * fz);
  wv[2] = (_Float16)(win * wx0 * fy  * wz0);
  wv[3] = (_Float16)(win * wx0 * fy  * fz);
  wv[4] = (_Float16)(win * fx  * wy0 * wz0);
  wv[5] = (_Float16)(win * fx  * wy0 * fz);
  wv[6] = (_Float16)(win * fx  * fy  * wz0);
  wv[7] = (_Float16)(win * fx  * fy  * fz);
  if (win == 0.f) base = 0xFFFFFFFFu;
  pairW[i]    = wv;
  pairCell[i] = base;
}

// ---------------------------------------------------------------------------
// Kernel 2: f32 feature [n][scin] -> f16 padded [n][cinp]
// ---------------------------------------------------------------------------
__global__ __launch_bounds__(256) void featcvt_kernel(
    const float* __restrict__ src, _Float16* __restrict__ dst,
    int n, int scin, int cinp)
{
  int idx = blockIdx.x * 256 + threadIdx.x;
  if (idx >= n * cinp) return;
  int p = idx / cinp, c = idx - p * cinp;
  dst[idx] = (_Float16)(c < scin ? src[p * scin + c] : 0.f);
}

// ---------------------------------------------------------------------------
// Kernel 3: build transposed/padded f16 weight matrix WT[cout][Kpad]
// ---------------------------------------------------------------------------
__global__ __launch_bounds__(256) void prep_wt_kernel(
    _Float16* __restrict__ WT, int Kpad, int CINP, int cin, int cinId,
    int coutValid, const float* __restrict__ Wc, int WcLd,
    const float* __restrict__ Wd, int WdLd,
    const float* __restrict__ bc, const float* __restrict__ bd,
    int addIdentity, int total)
{
  int idx = blockIdx.x * 256 + threadIdx.x;
  if (idx >= total) return;
  int n = idx / Kpad, k = idx - n * Kpad;
  float v = 0.f;
  int KCELLS = 64 * CINP;
  if (n < coutValid) {
    if (k < KCELLS) {
      int cell = k / CINP, c = k - cell * CINP;
      if (Wc && c < cin) v = Wc[(cell * cin + c) * WcLd + n];
    } else if (k < KCELLS + CINP) {
      int c = k - KCELLS;
      if (Wd && c < cinId) v = Wd[c * WdLd + n];
      if (addIdentity && c == n) v += 1.f;
    } else if (k == KCELLS + CINP) {
      if (bc) v += bc[n];
      if (bd) v += bd[n];
    }
  }
  WT[idx] = (_Float16)v;
}

// ---------------------------------------------------------------------------
// Kernel 4: fused cconv layer for a tile of 16 output particles.
//   Phase 0: async global->LDS staging of pair records (ASYNCcnt path)
//   Phase A: scatter 64 nbrs x 8 corners into f16 A[16][KPAD] in LDS using
//            wide collision-free read-modify-writes
//   Phase B: D = A x WT^T via v_wmma_f32_16x16x32_f16; compile-time split-K
//            across all 8 waves, partials reduced through LDS scratch
// ---------------------------------------------------------------------------
template<int CINP, int NT>
__global__ __launch_bounds__(256) void cconv_kernel(
    const v8h* __restrict__ pairW, const unsigned* __restrict__ pairCell,
    const int* __restrict__ nbr,
    const _Float16* __restrict__ featIn,   // [n_in][CINP]
    const _Float16* __restrict__ featOut,  // [16000][CINP] or null
    const _Float16* __restrict__ WT,       // [cout_pad][KPAD]
    int reluMode,                          // 0 none, 1 all, 2 cols<32
    _Float16* __restrict__ outH, int outStride, int outBase,
    float* __restrict__ outF, int outFCols)
{
  constexpr int KCELLS = 64 * CINP;
  constexpr int KPAD   = kpad_of(CINP);
  constexpr int KPAD32 = KPAD / 32;
  constexpr int NS     = 8 / NT;           // split-K factor (8 waves total)
  constexpr int TRIPS  = KPAD32 / NS;      // exact: KPAD32 divisible by 8
  constexpr int G      = (CINP >= 8) ? 8 : 4;
  constexpr int CPG    = CINP / G;
  using vG = typename HVec<G>::T;

  extern __shared__ __align__(32) char smem[];
  _Float16* Ah    = (_Float16*)smem;                    // 16*KPAD halves
  v8h*      sPw   = (v8h*)(smem + (size_t)16 * KPAD * 2);
  unsigned* sCell = (unsigned*)(sPw + 1024);
  int*      sNbr  = (int*)(sCell + 1024);
  v8h*      A8    = (v8h*)Ah;

  const int tid = threadIdx.x;

  // zero A tile (128-bit LDS stores)
  v8h z8 = {};
  for (int i = tid; i < 2 * KPAD; i += 256) A8[i] = z8;

  // Phase 0: stage this tile's 1024 pair records via async global->LDS DMA
  const int pbase = blockIdx.x * 1024;
  for (int i = tid; i < 1024; i += 256) {
    unsigned ldsW = (unsigned)(uintptr_t)(sPw + i);
    unsigned ldsC = (unsigned)(uintptr_t)(sCell + i);
    unsigned ldsN = (unsigned)(uintptr_t)(sNbr + i);
    const void* gW = (const void*)(pairW + pbase + i);
    const void* gC = (const void*)(pairCell + pbase + i);
    const void* gN = (const void*)(nbr + pbase + i);
    asm volatile("global_load_async_to_lds_b128 %0, %1, off"
                 :: "v"(ldsW), "v"(gW) : "memory");
    asm volatile("global_load_async_to_lds_b32 %0, %1, off"
                 :: "v"(ldsC), "v"(gC) : "memory");
    asm volatile("global_load_async_to_lds_b32 %0, %1, off"
                 :: "v"(ldsN), "v"(gN) : "memory");
  }
  asm volatile("s_wait_asynccnt 0x0" ::: "memory");
  __syncthreads();

  // Phase A: scatter-accumulate, one thread owns (particle, G-channel group)
  for (int idx = tid; idx < 16 * CPG; idx += 256) {
    int p = idx / CPG, g = idx - (idx / CPG) * CPG;
    _Float16* ArowH = Ah + (size_t)p * KPAD;
    int prow = p * 64;
    for (int k = 0; k < 64; ++k) {
      unsigned base = sCell[prow + k];
      if (base == 0xFFFFFFFFu) continue;
      int nb = sNbr[prow + k];
      vG fp = ((const vG*)(featIn + (size_t)nb * CINP))[g];
      v8h w = sPw[prow + k];
#pragma unroll
      for (int j = 0; j < 8; ++j) {
        const int coff = ((j >> 2) & 1) * 16 + ((j >> 1) & 1) * 4 + (j & 1);
        _Float16 wj = w[j];
        vG* addr = ((vG*)(ArowH + (base + coff) * CINP)) + g;
        *addr = *addr + fp * wj;
      }
    }
    // identity block (dense path / residual input)
    int gp = blockIdx.x * 16 + p;
    if (featOut)
      ((vG*)(ArowH + KCELLS))[g] = ((const vG*)(featOut + (size_t)gp * CINP))[g];
  }
  if (tid < 16) Ah[(size_t)tid * KPAD + KCELLS + CINP] = (_Float16)1.0f; // bias
  __syncthreads();

  // Phase B: WMMA, split-K across all 8 waves. wave = ksplit*NT + ntile.
  // readfirstlane makes the decomposition SGPR-resident -> scalar control flow.
  const int wave   = __builtin_amdgcn_readfirstlane(tid >> 5);
  const int lane   = tid & 31;
  const int ntile  = wave % NT;
  const int ksplit = wave / NT;
  const int mrow = lane & 15;        // M for A-frag, N for B-frag
  const int hsel = lane >> 4;        // lane-half K selector per ISA layout
  const int ncol = mrow + ntile * 16;
  const v8h* Arow8 = (const v8h*)(Ah + (size_t)mrow * KPAD);
  const v8h* Brow8 = (const v8h*)(WT + (size_t)ncol * KPAD);
  v8f acc = {};
#pragma unroll 4
  for (int i = 0; i < TRIPS; ++i) {
    int kk = ksplit + i * NS;
    v8h alo = Arow8[4 * kk + hsel];          // K = k0 + 8*hsel .. +7
    v8h ahi = Arow8[4 * kk + 2 + hsel];      // K = k0+16+8*hsel .. +7
    v8h blo = Brow8[4 * kk + hsel];
    v8h bhi = Brow8[4 * kk + 2 + hsel];
    v16h a = __builtin_shufflevector(alo, ahi, 0,1,2,3,4,5,6,7,8,9,10,11,12,13,14,15);
    v16h b = __builtin_shufflevector(blo, bhi, 0,1,2,3,4,5,6,7,8,9,10,11,12,13,14,15);
    acc = __builtin_amdgcn_wmma_f32_16x16x32_f16(false, a, false, b, (short)0,
                                                 acc, false, false);
  }
  // reduce split-K partials through LDS scratch (reuse dead pair staging area)
  v8f* sPart = (v8f*)sPw;                       // 8 waves * 32 lanes * 32B = 8KB
  if (ksplit > 0) sPart[wave * 32 + lane] = acc;
  __syncthreads();
  if (ksplit == 0) {
#pragma unroll
    for (int s = 1; s < NS; ++s)
      acc = acc + sPart[(ntile + s * NT) * 32 + lane];
    // Epilogue: C/D layout -> lane covers (M = r + 8*hsel, N = lane&15)
    int gbase = blockIdx.x * 16;
#pragma unroll
    for (int r = 0; r < 8; ++r) {
      int   m = r + (hsel << 3);
      float v = acc[r];
      if (reluMode == 1 || (reluMode == 2 && ncol < 32)) v = fmaxf(v, 0.f);
      int g = gbase + m;
      if (outH) outH[(size_t)g * outStride + outBase + ncol] = (_Float16)v;
      if (outF && ncol < outFCols) outF[(size_t)g * outFCols + ncol] = v;
    }
  }
}

// ---------------------------------------------------------------------------
// Host driver
// ---------------------------------------------------------------------------
static inline size_t alignup(size_t x) { return (x + 255) & ~(size_t)255; }

extern "C" void kernel_launch(void* const* d_in, const int* in_sizes, int n_in,
                              void* d_out, int out_size, void* d_ws, size_t ws_size,
                              hipStream_t stream) {
  (void)in_sizes; (void)n_in; (void)out_size; (void)ws_size;
  const float* fluid_pos = (const float*)d_in[0];
  const float* wall_pos  = (const float*)d_in[1];
  const float* fluid_vel = (const float*)d_in[2];
  const float* wall_nrm  = (const float*)d_in[3];
  const int*   nbr_wf    = (const int*)d_in[4];
  const unsigned char* mask_wf = (const unsigned char*)d_in[5];
  const int*   nbr_ff    = (const int*)d_in[6];
  const unsigned char* mask_ff = (const unsigned char*)d_in[7];
  const float* W_wall1  = (const float*)d_in[8];
  const float* b_wall1  = (const float*)d_in[9];
  const float* W_fluid1 = (const float*)d_in[10];
  const float* b_fluid1 = (const float*)d_in[11];
  const float* W_d1 = (const float*)d_in[12];
  const float* b_d1 = (const float*)d_in[13];
  const float* W_c2 = (const float*)d_in[14];
  const float* b_c2 = (const float*)d_in[15];
  const float* W_d2 = (const float*)d_in[16];
  const float* b_d2 = (const float*)d_in[17];
  const float* W_c3 = (const float*)d_in[18];
  const float* b_c3 = (const float*)d_in[19];
  const float* W_d3 = (const float*)d_in[20];
  const float* b_d3 = (const float*)d_in[21];
  const float* W_c4 = (const float*)d_in[22];
  const float* b_c4 = (const float*)d_in[23];
  const float* W_d4 = (const float*)d_in[24];
  const float* b_d4 = (const float*)d_in[25];

  // --- workspace carve-out ---
  char* ws = (char*)d_ws;
  size_t off = 0;
  auto take = [&](size_t bytes) { char* p = ws + off; off += alignup(bytes); return p; };
  v8h*      pairW_ff = (v8h*)     take((size_t)NPAIRS * 16);
  unsigned* pairC_ff = (unsigned*)take((size_t)NPAIRS * 4);
  v8h*      pairW_wf = (v8h*)     take((size_t)NPAIRS * 16);
  unsigned* pairC_wf = (unsigned*)take((size_t)NPAIRS * 4);
  _Float16* featVel  = (_Float16*)take((size_t)N_FLUID * 4 * 2);
  _Float16* featWall = (_Float16*)take((size_t)N_WALL * 4 * 2);
  _Float16* out1     = (_Float16*)take((size_t)N_FLUID * 96 * 2);
  _Float16* out2     = (_Float16*)take((size_t)N_FLUID * 64 * 2);
  _Float16* out3     = (_Float16*)take((size_t)N_FLUID * 64 * 2);
  const int KP1 = kpad_of(4), KP2 = kpad_of(96), KP3 = kpad_of(64);
  _Float16* WT1w = (_Float16*)take((size_t)32 * KP1 * 2);
  _Float16* WT1f = (_Float16*)take((size_t)64 * KP1 * 2);
  _Float16* WT2  = (_Float16*)take((size_t)64 * KP2 * 2);
  _Float16* WT3  = (_Float16*)take((size_t)64 * KP3 * 2);
  _Float16* WT4  = (_Float16*)take((size_t)16 * KP3 * 2);

  const int smem4  = 16 * KP1 * 2 + 1024 * (16 + 4 + 4);
  const int smem96 = 16 * KP2 * 2 + 1024 * (16 + 4 + 4);
  const int smem64 = 16 * KP3 * 2 + 1024 * (16 + 4 + 4);
  (void)hipFuncSetAttribute(reinterpret_cast<const void*>(cconv_kernel<4, 2>),
                            hipFuncAttributeMaxDynamicSharedMemorySize, smem4);
  (void)hipFuncSetAttribute(reinterpret_cast<const void*>(cconv_kernel<4, 4>),
                            hipFuncAttributeMaxDynamicSharedMemorySize, smem4);
  (void)hipFuncSetAttribute(reinterpret_cast<const void*>(cconv_kernel<96, 4>),
                            hipFuncAttributeMaxDynamicSharedMemorySize, smem96);
  (void)hipFuncSetAttribute(reinterpret_cast<const void*>(cconv_kernel<64, 4>),
                            hipFuncAttributeMaxDynamicSharedMemorySize, smem64);
  (void)hipFuncSetAttribute(reinterpret_cast<const void*>(cconv_kernel<64, 1>),
                            hipFuncAttributeMaxDynamicSharedMemorySize, smem64);

  // --- geometry (reused by all layers) ---
  int gpair = (NPAIRS + 255) / 256;
  geom_kernel<<<gpair, 256, 0, stream>>>(fluid_pos, fluid_pos, nbr_ff, mask_ff,
                                         NPAIRS, pairW_ff, pairC_ff);
  geom_kernel<<<gpair, 256, 0, stream>>>(wall_pos, fluid_pos, nbr_wf, mask_wf,
                                         NPAIRS, pairW_wf, pairC_wf);
  // --- f16 feature conversion ---
  featcvt_kernel<<<(N_FLUID * 4 + 255) / 256, 256, 0, stream>>>(fluid_vel, featVel,
                                                                N_FLUID, 3, 4);
  featcvt_kernel<<<(N_WALL * 4 + 255) / 256, 256, 0, stream>>>(wall_nrm, featWall,
                                                               N_WALL, 3, 4);
  // --- weight prep ---
  auto prep = [&](_Float16* wt, int kp, int cinp, int cin, int cinId, int coutLocal,
                  int coutValid, const float* Wc, int WcLd, const float* Wd, int WdLd,
                  const float* bc, const float* bd, int addId) {
    int total = coutLocal * kp;
    prep_wt_kernel<<<(total + 255) / 256, 256, 0, stream>>>(
        wt, kp, cinp, cin, cinId, coutValid, Wc, WcLd, Wd, WdLd, bc, bd, addId, total);
  };
  prep(WT1w, KP1, 4, 3, 3, 32, 32, W_wall1, 32, nullptr, 0, b_wall1, nullptr, 0);
  prep(WT1f, KP1, 4, 3, 3, 32, 32, W_fluid1, 32, nullptr, 0, b_fluid1, nullptr, 0);
  prep(WT1f + (size_t)32 * KP1, KP1, 4, 3, 3, 32, 32, nullptr, 0, W_d1, 32,
       b_d1, nullptr, 0);
  prep(WT2, KP2, 96, 96, 96, 64, 64, W_c2, 64, W_d2, 64, b_c2, b_d2, 0);
  prep(WT3, KP3, 64, 64, 64, 64, 64, W_c3, 64, W_d3, 64, b_c3, b_d3, 1);
  prep(WT4, KP3, 64, 64, 64, 16, 3, W_c4, 3, W_d4, 3, b_c4, b_d4, 0);

  // --- layers (1000 tiles of 16 particles) ---
  dim3 grid(N_FLUID / 16), blk(256);
  // L1 wall branch -> out1[:, 0:32], relu all
  cconv_kernel<4, 2><<<grid, blk, smem4, stream>>>(
      pairW_wf, pairC_wf, nbr_wf, featWall, nullptr, WT1w,
      1, out1, 96, 0, nullptr, 0);
  // L1 fluid cconv (relu) + dense (no relu) -> out1[:, 32:96]
  cconv_kernel<4, 4><<<grid, blk, smem4, stream>>>(
      pairW_ff, pairC_ff, nbr_ff, featVel, featVel, WT1f,
      2, out1, 96, 32, nullptr, 0);
  // L2: relu(cconv(out1) + out1 @ Wd2 + b)
  cconv_kernel<96, 4><<<grid, blk, smem96, stream>>>(
      pairW_ff, pairC_ff, nbr_ff, out1, out1, WT2,
      1, out2, 64, 0, nullptr, 0);
  // L3: relu(out2 + cconv(out2) + out2 @ Wd3 + b)  (identity folded into WT3)
  cconv_kernel<64, 4><<<grid, blk, smem64, stream>>>(
      pairW_ff, pairC_ff, nbr_ff, out2, out2, WT3,
      1, out3, 64, 0, nullptr, 0);
  // L4: cconv(out3) + out3 @ Wd4 + b -> f32 d_out [16000,3]
  cconv_kernel<64, 1><<<grid, blk, smem64, stream>>>(
      pairW_ff, pairC_ff, nbr_ff, out3, out3, WT4,
      0, nullptr, 0, 0, (float*)d_out, 3);
}
